// XDCNSemSegFPNHead_53506702573876
// MI455X (gfx1250) — matrix operand order repaired and checked by reading, not compiled, not run on hardware
//
#include <hip/hip_runtime.h>

// ---------------------------------------------------------------------------
// CDNA5 (gfx1250) WMMA types
// ---------------------------------------------------------------------------
typedef __attribute__((ext_vector_type(16))) _Float16 v16h;
typedef __attribute__((ext_vector_type(8)))  _Float16 v8h;
typedef __attribute__((ext_vector_type(8)))  float    v8f;

#define TM 64   // output-channel tile per block (16 per wave x 4 waves)
#define TN 64   // pixel tile per block
#define TK 32   // K chunk (WMMA f16 K)

// ---------------------------------------------------------------------------
// Fragment loaders for 16-bit A(16x32)/B(32x16), contiguous K order (same for A and B).
// A: lane<16 holds M=lane, K = 0..15; lane>=16 same M, K = 16..31.
// B: lane<16 holds N=lane, K = 0..15 (2 per VGPR); lane>=16 K = 16..31.
// LDS is row-major [row][K] so (K, K+1) pairs are contiguous -> ds_load_b128.
// ---------------------------------------------------------------------------
__device__ __forceinline__ v16h load_a_frag(const _Float16* Alds, int row, int half) {
  v16h a;
#pragma unroll
  for (int j = 0; j < 16; ++j) {
    const int v = j >> 1, p = j & 1;
    const int kb = (v < 4) ? (v * 2 + p) : (16 + (v - 4) * 2 + p);
    a[j] = Alds[row * TK + half * 16 + j];
  }
  return a;
}

__device__ __forceinline__ v16h load_b_frag(const _Float16* Blds, int col, int half) {
  v16h b;
#pragma unroll
  for (int j = 0; j < 16; ++j) {
    const int v = j >> 1, p = j & 1;
    b[j] = Blds[col * TK + half * 16 + v * 2 + p];
  }
  return b;
}

// ---------------------------------------------------------------------------
// Generic conv GEMM: y[co,p] = sum_{tap,c} w[co, c, tap] * patch(c, tap, p) + b
// KK == 9 -> 3x3 SAME (zero pad);  KK == 1 -> pointwise conv.
// K ordering: k = tap * C + c  (C is a multiple of 32 in all calls).
// Software-pipelined: chunk q+1 is staged in registers during chunk q's WMMAs.
// Fill decomposition: each thread owns row (tid&63) x contiguous K-run of 16,
// so the LDS commit is 4x ds_store_b128 per chunk.
// ---------------------------------------------------------------------------
__global__ __launch_bounds__(128)
void conv_im2col_wmma_kernel(const float* __restrict__ x,
                             const float* __restrict__ w,
                             const float* __restrict__ bias,
                             float* __restrict__ y,
                             int C, int Co, int H, int W, int KK)
{
  const int HW   = H * W;
  const int tid  = threadIdx.x;
  const int lane = tid & 31;
  const int wave = tid >> 5;
  const int half = lane >> 4;
  const int l16  = lane & 15;
  const int pixbase = blockIdx.x * TN;
  const int cob     = blockIdx.y * TM;
  const int cch = C >> 5;          // K-chunks per tap
  const int nq  = KK * cch;

  const int nB  = tid & 63;        // B column == A row owned by this thread
  const int grp = tid >> 6;        // K half-run: [grp*16, grp*16+16)
  const int pfx = pixbase + nB;
  const int yB  = pfx / W;         // hoisted out of the K loop
  const int xB  = pfx - yB * W;
  const int coA = cob + nB;        // A row's output channel (uniform per thread)
  const bool mok = (coA < Co);

  __shared__ alignas(16) _Float16 Alds[TM * TK];
  __shared__ alignas(16) _Float16 Blds[TN * TK];

  alignas(16) _Float16 ra[16];
  alignas(16) _Float16 rb[16];

  auto stage = [&](int tap, int cbase) {
    const int t3 = tap / 3;                       // division by constant
    const int dy = (KK == 9) ? (t3 - 1) : 0;
    const int dx = (KK == 9) ? (tap - t3 * 3 - 1) : 0;
    const int py = yB + dy, px = xB + dx;
    const bool inb = (py >= 0) && (py < H) && (px >= 0) && (px < W);
    const int  sof = inb ? (py * W + px) : 0;
    const int  kb  = cbase + grp * 16;
    const float* xp = x + (size_t)kb * HW + sof;
    const float* wp = w + (size_t)coA * C * KK + (size_t)kb * KK + tap;
#pragma unroll
    for (int i = 0; i < 16; ++i) {
      ra[i] = (_Float16)(mok ? wp[i * KK] : 0.0f);
      rb[i] = (_Float16)(inb ? xp[(size_t)i * HW] : 0.0f);
    }
  };

  v8f acc[4] = {};

  int tap = 0, cb = 0;
  stage(0, 0);
  for (int q = 0; q < nq; ++q) {
    // commit staged tiles to LDS: 4x ds_store_b128
    {
      v8h* A0 = (v8h*)&Alds[nB * TK + grp * 16];
      A0[0] = *(const v8h*)&ra[0];
      A0[1] = *(const v8h*)&ra[8];
      v8h* B0 = (v8h*)&Blds[nB * TK + grp * 16];
      B0[0] = *(const v8h*)&rb[0];
      B0[1] = *(const v8h*)&rb[8];
    }
    __syncthreads();

    // stage next chunk's global loads while WMMAs consume LDS
    int ntap = tap, ncb = cb + 1;
    if (ncb == cch) { ncb = 0; ++ntap; }
    if (q + 1 < nq) stage(ntap, ncb * 32);
    tap = ntap; cb = ncb;

    const v16h a = load_a_frag(Alds, wave * 16 + l16, half);
#pragma unroll
    for (int f = 0; f < 4; ++f) {
      const v16h b = load_b_frag(Blds, f * 16 + l16, half);
      acc[f] = __builtin_amdgcn_wmma_f32_16x16x32_f16(
          false, a, false, b, (short)0, acc[f], false, false);
    }
    __syncthreads();
  }

  // store (C/D layout: VGPR r -> M = half*8 + r, col = lane%16)
#pragma unroll
  for (int f = 0; f < 4; ++f) {
#pragma unroll
    for (int r = 0; r < 8; ++r) {
      const int co = cob + wave * 16 + half * 8 + r;
      if (co < Co) {
        const int p = pixbase + f * 16 + l16;
        y[(size_t)co * HW + p] = acc[f][r] + bias[co];
      }
    }
  }
}

// ---------------------------------------------------------------------------
// Deformable conv GEMM (DCNv1, 3x3, pad 1, 1 deform group).
// Per-block precompute of bilinear metadata (4 idx + 4 weights per pixel,tap),
// shared across all channels / K chunks. Same register-staged pipeline.
// ---------------------------------------------------------------------------
__global__ __launch_bounds__(128)
void deform_conv_wmma_kernel(const float* __restrict__ x,
                             const float* __restrict__ off,
                             const float* __restrict__ dwt,
                             float* __restrict__ y,
                             int C, int Co, int H, int W)
{
  const int HW   = H * W;
  const int tid  = threadIdx.x;
  const int lane = tid & 31;
  const int wave = tid >> 5;
  const int half = lane >> 4;
  const int l16  = lane & 15;
  const int pixbase = blockIdx.x * TN;
  const int cob     = blockIdx.y * TM;
  const int cch = C >> 5;
  const int nq  = 9 * cch;

  const int nB  = tid & 63;
  const int grp = tid >> 6;
  const int coA = cob + nB;
  const bool mok = (coA < Co);

  __shared__ alignas(16) _Float16 Alds[TM * TK];
  __shared__ alignas(16) _Float16 Blds[TN * TK];
  __shared__ alignas(16) int   sIdx[TN * 9 * 4];
  __shared__ alignas(16) float sWgt[TN * 9 * 4];

  // ---- bilinear sampling metadata: 64 pixels x 9 taps ----
  for (int e = tid; e < TN * 9; e += 128) {
    const int n = e / 9, t = e - n * 9;
    const int p = pixbase + n;
    const int yy = p / W, xx = p - yy * W;
    const float dyo = off[(size_t)(2 * t)     * HW + p];
    const float dxo = off[(size_t)(2 * t + 1) * HW + p];
    const float py = (float)yy + (float)(t / 3 - 1) + dyo;
    const float px = (float)xx + (float)(t % 3 - 1) + dxo;
    const float y0f = floorf(py), x0f = floorf(px);
    const float wy = py - y0f, wx = px - x0f;
    const int y0 = (int)y0f, x0 = (int)x0f;
#pragma unroll
    for (int j = 0; j < 4; ++j) {
      const int yi = y0 + (j >> 1);
      const int xi = x0 + (j & 1);
      const bool valid = (yi >= 0) && (yi <= H - 1) && (xi >= 0) && (xi <= W - 1);
      const int yc = min(max(yi, 0), H - 1);
      const int xc = min(max(xi, 0), W - 1);
      const float ww = ((j >> 1) ? wy : 1.0f - wy) * ((j & 1) ? wx : 1.0f - wx);
      sIdx[e * 4 + j] = yc * W + xc;
      sWgt[e * 4 + j] = valid ? ww : 0.0f;
    }
  }
  __syncthreads();   // meta must be visible before first stage()

  alignas(16) _Float16 ra[16];
  alignas(16) _Float16 rb[16];

  auto stage = [&](int tap, int cbase) {
    const int mi = (nB * 9 + tap) * 4;           // 16B-aligned -> ds_load_b128
    const int4   iv = *(const int4*)&sIdx[mi];
    const float4 wv = *(const float4*)&sWgt[mi];
    const int kb = cbase + grp * 16;
    const float* xp = x + (size_t)kb * HW;
    const float* wp = dwt + (size_t)coA * C * 9 + (size_t)kb * 9 + tap;
#pragma unroll
    for (int i = 0; i < 16; ++i) {
      ra[i] = (_Float16)(mok ? wp[i * 9] : 0.0f);
      const float* pc = xp + (size_t)i * HW;
      const float v = wv.x * pc[iv.x] + wv.y * pc[iv.y]
                    + wv.z * pc[iv.z] + wv.w * pc[iv.w];
      rb[i] = (_Float16)v;
    }
  };

  v8f acc[4] = {};

  int tap = 0, cb = 0;
  stage(0, 0);
  for (int q = 0; q < nq; ++q) {
    {
      v8h* A0 = (v8h*)&Alds[nB * TK + grp * 16];
      A0[0] = *(const v8h*)&ra[0];
      A0[1] = *(const v8h*)&ra[8];
      v8h* B0 = (v8h*)&Blds[nB * TK + grp * 16];
      B0[0] = *(const v8h*)&rb[0];
      B0[1] = *(const v8h*)&rb[8];
    }
    __syncthreads();

    int ntap = tap, ncb = cb + 1;
    if (ncb == cch) { ncb = 0; ++ntap; }
    if (q + 1 < nq) stage(ntap, ncb * 32);
    tap = ntap; cb = ncb;

    const v16h a = load_a_frag(Alds, wave * 16 + l16, half);
#pragma unroll
    for (int f = 0; f < 4; ++f) {
      const v16h b = load_b_frag(Blds, f * 16 + l16, half);
      acc[f] = __builtin_amdgcn_wmma_f32_16x16x32_f16(
          false, a, false, b, (short)0, acc[f], false, false);
    }
    __syncthreads();
  }

#pragma unroll
  for (int f = 0; f < 4; ++f) {
#pragma unroll
    for (int r = 0; r < 8; ++r) {
      const int co = cob + wave * 16 + half * 8 + r;
      if (co < Co) {
        const int p = pixbase + f * 16 + l16;
        y[(size_t)co * HW + p] = acc[f][r];
      }
    }
  }
}

// ---------------------------------------------------------------------------
// GroupNorm (32 groups) + ReLU, in place. One block per group.
// ---------------------------------------------------------------------------
__global__ __launch_bounds__(256)
void groupnorm_relu_kernel(float* __restrict__ y,
                           const float* __restrict__ gamma,
                           const float* __restrict__ beta,
                             int cpg, int HW)
{
  __shared__ float ssum[256];
  __shared__ float ssq[256];
  const int g = blockIdx.x, tid = threadIdx.x;
  const size_t base = (size_t)g * cpg * HW;
  const int n = cpg * HW;
  float s = 0.0f, s2 = 0.0f;
  for (int i = tid; i < n; i += 256) {
    const float v = y[base + i];
    s += v; s2 += v * v;
  }
  ssum[tid] = s; ssq[tid] = s2;
  __syncthreads();
  for (int st = 128; st > 0; st >>= 1) {
    if (tid < st) { ssum[tid] += ssum[tid + st]; ssq[tid] += ssq[tid + st]; }
    __syncthreads();
  }
  const float mean = ssum[0] / (float)n;
  const float var  = ssq[0] / (float)n - mean * mean;
  const float rstd = rsqrtf(var + 1e-5f);
  for (int i = tid; i < n; i += 256) {
    const int c = g * cpg + i / HW;
    float v = (y[base + i] - mean) * rstd * gamma[c] + beta[c];
    y[base + i] = v > 0.0f ? v : 0.0f;
  }
}

// ---------------------------------------------------------------------------
// Bilinear upsample (align_corners=False), integer scale (1/2/4/8).
// Writes into `out` at channel offset cOff with channel stride outCstride.
// ---------------------------------------------------------------------------
__global__ __launch_bounds__(256)
void upsample_bilinear_kernel(const float* __restrict__ in, float* __restrict__ out,
                              int C, int H, int W, int scale,
                              int cOff, int outCstride)
{
  const int Ho = H * scale, Wo = W * scale;
  const long long total = (long long)C * Ho * Wo;
  const long long idx = (long long)blockIdx.x * 256 + threadIdx.x;
  if (idx >= total) return;
  const int c   = (int)(idx / ((long long)Ho * Wo));
  const int rem = (int)(idx - (long long)c * Ho * Wo);
  const int yo = rem / Wo, xo = rem - yo * Wo;
  const float inv = 1.0f / (float)scale;
  float sy = ((float)yo + 0.5f) * inv - 0.5f;
  sy = fminf(fmaxf(sy, 0.0f), (float)(H - 1));
  const int y0 = (int)sy; const float fy = sy - (float)y0;
  const int y1 = min(y0 + 1, H - 1);
  float sx = ((float)xo + 0.5f) * inv - 0.5f;
  sx = fminf(fmaxf(sx, 0.0f), (float)(W - 1));
  const int x0 = (int)sx; const float fx = sx - (float)x0;
  const int x1 = min(x0 + 1, W - 1);
  const float* p = in + (size_t)c * H * W;
  const float v = (p[y0 * W + x0] * (1.0f - fy) + p[y1 * W + x0] * fy) * (1.0f - fx)
                + (p[y0 * W + x1] * (1.0f - fy) + p[y1 * W + x1] * fy) * fx;
  out[(size_t)(cOff + c) * outCstride + rem] = v;
}

// ---------------------------------------------------------------------------
// Host-side orchestration (graph-capture safe: only kernel launches on stream)
// ---------------------------------------------------------------------------
extern "C" void kernel_launch(void* const* d_in, const int* in_sizes, int n_in,
                              void* d_out, int out_size, void* d_ws, size_t ws_size,
                              hipStream_t stream) {
  (void)in_sizes; (void)n_in; (void)out_size; (void)ws_size;

  const float* P[4]  = {(const float*)d_in[0], (const float*)d_in[1],
                        (const float*)d_in[2], (const float*)d_in[3]};
  const float* ow[3] = {(const float*)d_in[4],  (const float*)d_in[9],  (const float*)d_in[14]};
  const float* ob[3] = {(const float*)d_in[5],  (const float*)d_in[10], (const float*)d_in[15]};
  const float* dw[3] = {(const float*)d_in[6],  (const float*)d_in[11], (const float*)d_in[16]};
  const float* gg[3] = {(const float*)d_in[7],  (const float*)d_in[12], (const float*)d_in[17]};
  const float* bb[3] = {(const float*)d_in[8],  (const float*)d_in[13], (const float*)d_in[18]};
  const float* pw = (const float*)d_in[19];
  const float* pb = (const float*)d_in[20];

  // workspace carve-up (bytes)
  const size_t SZ_ACT = (size_t)256 * 24576 * 4;       // 25,165,824
  const size_t SZ_OFF = (size_t)2097152;               // >= 18*24576*4
  const size_t SZ_CAT = (size_t)512 * 24576 * 4;       // 50,331,648
  char* ws = (char*)d_ws;
  float* bufA   = (float*)(ws);
  float* bufB   = (float*)(ws + SZ_ACT);
  float* offb   = (float*)(ws + 2 * SZ_ACT);
  float* cat    = (float*)(ws + 2 * SZ_ACT + SZ_OFF);
  float* scores = (float*)(ws + 2 * SZ_ACT + SZ_OFF + SZ_CAT);

  const int lc_in[3]  = {256, 256, 128};
  const int lc_out[3] = {256, 128, 128};

  for (int lvl = 0; lvl < 4; ++lvl) {
    const int H = 128 >> lvl, W = 192 >> lvl, HW = H * W;
    const float* cur = P[lvl];
    for (int li = 0; li < 3; ++li) {
      const int C   = lc_in[li];
      const int Cou = lc_out[li];
      float* dst = (li == 1) ? bufB : bufA;   // L0->A, L1->B, L2->A (ping-pong)

      dim3 gc(HW / TN, 1);                    // offset conv: Co=18 fits one M tile block
      conv_im2col_wmma_kernel<<<gc, 128, 0, stream>>>(cur, ow[li], ob[li], offb,
                                                      C, 18, H, W, 9);
      dim3 gd(HW / TN, (Cou + TM - 1) / TM);
      deform_conv_wmma_kernel<<<gd, 128, 0, stream>>>(cur, offb, dw[li], dst,
                                                      C, Cou, H, W);
      groupnorm_relu_kernel<<<32, 256, 0, stream>>>(dst, gg[li], bb[li],
                                                    Cou / 32, HW);
      cur = dst;
    }
    const int scale = 1 << lvl;
    const long long tot = (long long)128 * 128 * 192;
    upsample_bilinear_kernel<<<(unsigned)((tot + 255) / 256), 256, 0, stream>>>(
        cur, cat, 128, H, W, scale, lvl * 128, 128 * 192);
  }

  // predictor 1x1 conv: [54, 512] x cat[512, 24576]
  dim3 gp(24576 / TN, 1);
  conv_im2col_wmma_kernel<<<gp, 128, 0, stream>>>(cat, pw, pb, scores,
                                                  512, 54, 128, 192, 1);

  // final 4x bilinear upsample to d_out [54, 512, 768]
  const long long tot = (long long)54 * 512 * 768;
  upsample_bilinear_kernel<<<(unsigned)((tot + 255) / 256), 256, 0, stream>>>(
      scores, (float*)d_out, 54, 128, 192, 4, 0, 512 * 768);
}
